// MultiHeadAttention_57062935495212
// MI455X (gfx1250) — matrix operand order repaired and checked by reading, not compiled, not run on hardware
//
#include <hip/hip_runtime.h>

typedef __bf16 bf16_t;
typedef __attribute__((ext_vector_type(16))) __bf16 v16bf;
typedef __attribute__((ext_vector_type(8)))  __bf16 v8bf;
typedef __attribute__((ext_vector_type(8)))  float  v8f;

#define D_MODEL 1024
#define N_HEADS 16
#define D_K     64
#define BATCH   2
#define SEQ     2048
#define MROWS   (BATCH * SEQ)   // 4096

// ---------------------------------------------------------------------------
// fp32 -> bf16 elementwise cast (grid-stride)
// ---------------------------------------------------------------------------
__global__ void cast_f32_bf16(const float* __restrict__ in,
                              bf16_t* __restrict__ out, int n) {
  int i = blockIdx.x * blockDim.x + threadIdx.x;
  int stride = gridDim.x * blockDim.x;
  for (; i < n; i += stride) out[i] = (bf16_t)in[i];
}

// ---------------------------------------------------------------------------
// WMMA fragment loaders (wave32 layouts per CDNA5 ISA 7.12.2).
// A 16x32 tile: lane m (lane%16); elems 0..7  = A[m][k0 + 8*hl .. +7]
//                                 elems 8..15 = A[m][k0+16+8*hl .. +7]
// B 32x16 tile: lane n (lane%16); elems 0..7  = B[n][k0 + 8*hl .. +7], elems 8..15 = B[n][k0+16+8*hl .. +7]
// (row pointers passed already offset by r*ld)
// ---------------------------------------------------------------------------
__device__ inline v16bf load_afrag(const bf16_t* __restrict__ arow, int k0, int hl) {
  v8bf a0 = *(const v8bf*)(arow + k0 + hl * 8);
  v8bf a1 = *(const v8bf*)(arow + k0 + 16 + hl * 8);
  return __builtin_shufflevector(a0, a1, 0,1,2,3,4,5,6,7,8,9,10,11,12,13,14,15);
}
__device__ inline v16bf load_bfrag(const bf16_t* __restrict__ brow, int k0, int hl) {
  v8bf b0 = *(const v8bf*)(brow + k0 + hl * 8);
  v8bf b1 = *(const v8bf*)(brow + k0 + 16 + hl * 8);
  return __builtin_shufflevector(b0, b1, 0,1,2,3,4,5,6,7,8,9,10,11,12,13,14,15);
}
__device__ inline v8f wmma_bf16(v16bf a, v16bf b, v8f c) {
  return __builtin_amdgcn_wmma_f32_16x16x32_bf16(false, a, false, b,
                                                 (short)0, c, false, false);
}

// ---------------------------------------------------------------------------
// Projection GEMM, 1x4 N-blocked: each wave computes C[16, 64] of
// X[M,K] @ W[N,K]^T (M=4096, N=K=1024), bf16 out.  Since D_K==64, an
// n-group is exactly one head: h = ng, d = j*16 + r.
// mode 0: store [b, h, s, d]    (Q, K)
// mode 1: store [b, h, d, s]    (V transposed for the context GEMM)
// ---------------------------------------------------------------------------
__global__ void proj_gemm(const bf16_t* __restrict__ X,
                          const bf16_t* __restrict__ W,
                          bf16_t* __restrict__ out, int mode) {
  const int wid = blockIdx.x * (blockDim.x >> 5) + (threadIdx.x >> 5);
  const int ngr = D_MODEL / 64;                 // 16 n-groups
  const int mt = wid / ngr, ng = wid % ngr;
  if (mt >= MROWS / 16) return;

  const int lane = threadIdx.x & 31;
  const int hl = lane >> 4, r = lane & 15;

  const bf16_t* arow = X + ((size_t)mt * 16 + r) * D_MODEL;
  const bf16_t* brow[4];
#pragma unroll
  for (int j = 0; j < 4; ++j)
    brow[j] = W + ((size_t)ng * 64 + j * 16 + r) * D_MODEL;

  v8f acc[4] = {{}, {}, {}, {}};
  for (int k0 = 0; k0 < D_MODEL; k0 += 32) {
    v16bf av = load_afrag(arow, k0, hl);
#pragma unroll
    for (int j = 0; j < 4; ++j)
      acc[j] = wmma_bf16(av, load_bfrag(brow[j], k0, hl), acc[j]);
  }

  const int h = ng;                             // head index (64 == D_K)
#pragma unroll
  for (int j = 0; j < 4; ++j) {
    const int d = j * 16 + r;
#pragma unroll
    for (int v = 0; v < 8; ++v) {
      const int m = mt * 16 + hl * 8 + v;
      const int b = m / SEQ, s = m % SEQ;
      const bf16_t val = (bf16_t)acc[j][v];
      if (mode == 0)
        out[(((size_t)b * N_HEADS + h) * SEQ + s) * D_K + d] = val;
      else
        out[(((size_t)b * N_HEADS + h) * D_K + d) * SEQ + s] = val;
    }
  }
}

// ---------------------------------------------------------------------------
// Scores, 1x4 K-pos-blocked: wave computes scores[16 q, 64 k] for one z,
// reusing the Q fragment across 4 key tiles.  Causal: skip group entirely
// when its first key tile is above the diagonal.
// ---------------------------------------------------------------------------
__global__ void scores_gemm(const bf16_t* __restrict__ Q,
                            const bf16_t* __restrict__ K,
                            float* __restrict__ attn) {
  const int z = blockIdx.z;
  const int wid = blockIdx.x * (blockDim.x >> 5) + (threadIdx.x >> 5);
  const int nkg = SEQ / 64;                     // 32 key groups
  const int qt = wid / nkg, kg = wid % nkg;
  if (qt >= SEQ / 16) return;
  if (kg * 4 > qt) return;                      // whole group masked

  const int lane = threadIdx.x & 31;
  const int hl = lane >> 4, r = lane & 15;

  const bf16_t* arow = Q + ((size_t)z * SEQ + qt * 16 + r) * D_K;
  const bf16_t* brow[4];
#pragma unroll
  for (int j = 0; j < 4; ++j)
    brow[j] = K + ((size_t)z * SEQ + kg * 64 + j * 16 + r) * D_K;

  v8f acc[4] = {{}, {}, {}, {}};
  for (int k0 = 0; k0 < D_K; k0 += 32) {
    v16bf av = load_afrag(arow, k0, hl);
#pragma unroll
    for (int j = 0; j < 4; ++j)
      acc[j] = wmma_bf16(av, load_bfrag(brow[j], k0, hl), acc[j]);
  }

#pragma unroll
  for (int j = 0; j < 4; ++j) {
    const int kp = kg * 64 + j * 16 + r;
#pragma unroll
    for (int v = 0; v < 8; ++v) {
      const int qp = qt * 16 + hl * 8 + v;
      attn[((size_t)z * SEQ + qp) * SEQ + kp] = acc[j][v] * 0.125f;
    }
  }
}

// ---------------------------------------------------------------------------
// Causal softmax, one 256-thread block per row (B*H*S rows).
// Writes exact 0 above the diagonal (matches exp(-1e9 - m) underflow).
// ---------------------------------------------------------------------------
__global__ void softmax_causal(float* __restrict__ attn) {
  const int row = blockIdx.x;                   // 0 .. B*H*S-1
  const int q = row % SEQ;
  float* p = attn + (size_t)row * SEQ;
  const int len = q + 1;
  const int tid = threadIdx.x;
  __shared__ float red[256];

  float m = -1e30f;
  for (int j = tid; j < len; j += 256) m = fmaxf(m, p[j]);
  red[tid] = m; __syncthreads();
  for (int s = 128; s > 0; s >>= 1) {
    if (tid < s) red[tid] = fmaxf(red[tid], red[tid + s]);
    __syncthreads();
  }
  m = red[0]; __syncthreads();

  float sum = 0.f;
  for (int j = tid; j < len; j += 256) sum += __expf(p[j] - m);
  red[tid] = sum; __syncthreads();
  for (int s = 128; s > 0; s >>= 1) {
    if (tid < s) red[tid] += red[tid + s];
    __syncthreads();
  }
  const float inv = 1.0f / red[0];

  for (int j = tid; j < SEQ; j += 256)
    p[j] = (j < len) ? __expf(p[j] - m) * inv : 0.f;
}

// ---------------------------------------------------------------------------
// Context, all 4 d-tiles per wave: ctx[b,s,h*64+d] = sum_k attn[z,s,k]*V[z,k,d].
// The fp32 attn fragment (the dominant HBM read of the whole problem) is
// loaded once per k-step and reused across the 4 WMMAs covering D_K.
// B = V_t [z, d, s] so B rows are k-contiguous.  K-loop stops at the causal
// boundary of the q-tile (attn is exact 0 beyond it).
// ---------------------------------------------------------------------------
__global__ void context_gemm(const float* __restrict__ attn,
                             const bf16_t* __restrict__ Vt,
                             bf16_t* __restrict__ ctx) {
  const int z = blockIdx.z;
  const int b = z / N_HEADS, h = z % N_HEADS;
  const int qt = blockIdx.x * (blockDim.x >> 5) + (threadIdx.x >> 5);
  if (qt >= SEQ / 16) return;

  const int lane = threadIdx.x & 31;
  const int hl = lane >> 4, r = lane & 15;

  const float* arow = attn + ((size_t)z * SEQ + qt * 16 + r) * SEQ;
  const bf16_t* brow[4];
#pragma unroll
  for (int j = 0; j < 4; ++j)
    brow[j] = Vt + ((size_t)z * D_K + j * 16 + r) * SEQ;

  const int kend = qt * 16 + 16;                // causal boundary, rounded up
  v8f acc[4] = {{}, {}, {}, {}};
  for (int k0 = 0; k0 < kend; k0 += 32) {
    v8f fa0 = *(const v8f*)(arow + k0 + hl * 8);
    v8f fa1 = *(const v8f*)(arow + k0 + 16 + hl * 8);
    v16bf av;
#pragma unroll
    for (int e = 0; e < 8; ++e) {
      av[e]     = (bf16_t)fa0[e];
      av[e + 8] = (bf16_t)fa1[e];
    }
#pragma unroll
    for (int j = 0; j < 4; ++j)
      acc[j] = wmma_bf16(av, load_bfrag(brow[j], k0, hl), acc[j]);
  }

#pragma unroll
  for (int j = 0; j < 4; ++j) {
#pragma unroll
    for (int v = 0; v < 8; ++v) {
      const int s = qt * 16 + hl * 8 + v;
      ctx[((size_t)b * SEQ + s) * D_MODEL + h * D_K + j * 16 + r] =
          (bf16_t)acc[j][v];
    }
  }
}

// ---------------------------------------------------------------------------
// Output GEMM, 1x4 N-blocked: out[M,N] = ctx[M,K] @ Wo[N,K]^T + b_o[N] (fp32)
// ---------------------------------------------------------------------------
__global__ void out_gemm(const bf16_t* __restrict__ X,
                         const bf16_t* __restrict__ W,
                         const float* __restrict__ bias,
                         float* __restrict__ out) {
  const int wid = blockIdx.x * (blockDim.x >> 5) + (threadIdx.x >> 5);
  const int ngr = D_MODEL / 64;
  const int mt = wid / ngr, ng = wid % ngr;
  if (mt >= MROWS / 16) return;

  const int lane = threadIdx.x & 31;
  const int hl = lane >> 4, r = lane & 15;

  const bf16_t* arow = X + ((size_t)mt * 16 + r) * D_MODEL;
  const bf16_t* brow[4];
#pragma unroll
  for (int j = 0; j < 4; ++j)
    brow[j] = W + ((size_t)ng * 64 + j * 16 + r) * D_MODEL;

  v8f acc[4] = {{}, {}, {}, {}};
  for (int k0 = 0; k0 < D_MODEL; k0 += 32) {
    v16bf av = load_afrag(arow, k0, hl);
#pragma unroll
    for (int j = 0; j < 4; ++j)
      acc[j] = wmma_bf16(av, load_bfrag(brow[j], k0, hl), acc[j]);
  }

#pragma unroll
  for (int j = 0; j < 4; ++j) {
    const int n = ng * 64 + j * 16 + r;
    const float bv = bias[n];
#pragma unroll
    for (int v = 0; v < 8; ++v) {
      const int m = mt * 16 + hl * 8 + v;
      out[(size_t)m * D_MODEL + n] = acc[j][v] + bv;
    }
  }
}

// ---------------------------------------------------------------------------
extern "C" void kernel_launch(void* const* d_in, const int* in_sizes, int n_in,
                              void* d_out, int out_size, void* d_ws, size_t ws_size,
                              hipStream_t stream) {
  const float* query = (const float*)d_in[0];
  const float* keyi  = (const float*)d_in[1];
  const float* value = (const float*)d_in[2];
  // d_in[3] = mask (causal; structure known statically, unused)
  const float* w_q = (const float*)d_in[4];
  const float* w_k = (const float*)d_in[5];
  const float* w_v = (const float*)d_in[6];
  const float* w_o = (const float*)d_in[7];
  const float* b_o = (const float*)d_in[8];

  float* out  = (float*)d_out;                              // [B,S,D]
  float* attn = out + (size_t)BATCH * SEQ * D_MODEL;        // [B,H,S,S]

  // workspace carve-up (~56 MB)
  char* ws = (char*)d_ws;
  size_t off = 0;
  auto take = [&](size_t bytes) -> char* {
    char* p = ws + off;
    off += (bytes + 255) & ~(size_t)255;
    return p;
  };
  const size_t actB = (size_t)MROWS * D_MODEL * sizeof(bf16_t);   // 8 MB
  const size_t wB   = (size_t)D_MODEL * D_MODEL * sizeof(bf16_t); // 2 MB
  bf16_t* xq  = (bf16_t*)take(actB);
  bf16_t* xk  = (bf16_t*)take(actB);
  bf16_t* xv  = (bf16_t*)take(actB);
  bf16_t* wqb = (bf16_t*)take(wB);
  bf16_t* wkb = (bf16_t*)take(wB);
  bf16_t* wvb = (bf16_t*)take(wB);
  bf16_t* wob = (bf16_t*)take(wB);
  bf16_t* qb  = (bf16_t*)take(actB);   // [b,h,s,d]
  bf16_t* kb  = (bf16_t*)take(actB);   // [b,h,s,d]
  bf16_t* vtb = (bf16_t*)take(actB);   // [b,h,d,s]
  bf16_t* ctx = xq;                    // reuse: xq dead after projections

  const int nAct = MROWS * D_MODEL;
  const int nW   = D_MODEL * D_MODEL;

  // 1) casts
  cast_f32_bf16<<<1024, 256, 0, stream>>>(query, xq, nAct);
  cast_f32_bf16<<<1024, 256, 0, stream>>>(keyi,  xk, nAct);
  cast_f32_bf16<<<1024, 256, 0, stream>>>(value, xv, nAct);
  cast_f32_bf16<<<512,  256, 0, stream>>>(w_q, wqb, nW);
  cast_f32_bf16<<<512,  256, 0, stream>>>(w_k, wkb, nW);
  cast_f32_bf16<<<512,  256, 0, stream>>>(w_v, wvb, nW);
  cast_f32_bf16<<<512,  256, 0, stream>>>(w_o, wob, nW);

  // 2) projections: (4096/16) * (1024/64) = 4096 waves, 8 waves/block
  proj_gemm<<<512, 256, 0, stream>>>(xq, wqb, qb,  0);
  proj_gemm<<<512, 256, 0, stream>>>(xk, wkb, kb,  0);
  proj_gemm<<<512, 256, 0, stream>>>(xv, wvb, vtb, 1);

  // 3) scores: per z, 128 q-tiles * 32 k-groups = 4096 waves -> 512 blocks
  scores_gemm<<<dim3(512, 1, BATCH * N_HEADS), 256, 0, stream>>>(qb, kb, attn);

  // 4) causal softmax: one block per row
  softmax_causal<<<BATCH * N_HEADS * SEQ, 256, 0, stream>>>(attn);

  // 5) context: per z, 128 q-tile waves -> 16 blocks
  context_gemm<<<dim3(16, 1, BATCH * N_HEADS), 256, 0, stream>>>(attn, vtb, ctx);

  // 6) output projection + bias
  out_gemm<<<512, 256, 0, stream>>>(ctx, wob, b_o, out);
}
